// ColorCNN_62654982914247
// MI455X (gfx1250) — matrix operand order repaired and checked by reading, not compiled, not run on hardware
//
#include <hip/hip_runtime.h>
#include <hip/hip_fp16.h>

typedef _Float16 v16h __attribute__((ext_vector_type(16)));
typedef float    v8f  __attribute__((ext_vector_type(8)));
typedef float    v2f  __attribute__((ext_vector_type(2)));

#define HW    50176      // 224*224
#define BATCH 16
#define HID   256
#define KOUT  64
#define W2STR 264        // padded halfs per W2 row (256 + 8) -> 528B, conflict-free b128
#define HSTR  264        // padded halfs per h row (per pixel)
#define MSTR  20         // padded floats per m row (per k)   -> 80B, conflict-free b128
#define NWAVE 8
#define BLK49 49         // blocks per batch image (49*8*128 = 50176)

// ---------------------------------------------------------------------------
// Kernel 1: per-pixel MLP (WMMA) + softmax + m store + deterministic num/den
// grid = 16*49 = 784 blocks, 256 threads (8 waves), each wave: 16 pix x 8 iters
// ---------------------------------------------------------------------------
__global__ __launch_bounds__(256) void colorcnn_main(
    const float* __restrict__ img, const float* __restrict__ W1,
    const float* __restrict__ b1,  const float* __restrict__ W2,
    float* __restrict__ m_out, float* __restrict__ partials)
{
  __shared__ float4   sW1[HID];                         // 4KB: [row] = (w0,w1,w2,0)
  __shared__ float    sB1[HID];                         // 1KB
  __shared__ _Float16 sW2[KOUT * W2STR];                // 33792B, rows padded
  __shared__ float    sAcc[NWAVE * 256];                // 8KB per-wave num/den accum
  __shared__ __align__(16) char sScr[NWAVE * 16 * HSTR * 2]; // 67584B per-wave h / m-stage

  const int t    = threadIdx.x;
  const int wv   = t >> 5;
  const int lid  = t & 31;
  const int lo   = (lid < 16) ? 1 : 0;
  const int pix  = lid & 15;
  const int half8 = lo ? 0 : 8;

  // ---- stage weights into LDS (once per block) ----
  // b1 via gfx1250 async global->LDS DMA (GVS mode), tracked by ASYNCcnt
  {
    unsigned ldsOff = (unsigned)(uintptr_t)(&sB1[t]);   // low 32 bits = LDS offset
    unsigned goff   = (unsigned)(t * 4);                // byte offset from SGPR base
    asm volatile("global_load_async_to_lds_b32 %0, %1, %2"
                 :: "v"(ldsOff), "v"(goff), "s"(b1) : "memory");
  }
  sW1[t] = make_float4(W1[t * 3 + 0], W1[t * 3 + 1], W1[t * 3 + 2], 0.0f);
  for (int idx = t; idx < KOUT * W2STR; idx += 256) {
    int k = idx / W2STR, hc = idx - k * W2STR;
    sW2[idx] = (hc < HID) ? (_Float16)W2[k * HID + hc] : (_Float16)0.0f;
  }
  for (int i = t; i < NWAVE * 256; i += 256) sAcc[i] = 0.0f;
  asm volatile("s_wait_asynccnt 0" ::: "memory");
  __syncthreads();

  const int g  = blockIdx.x;
  const int b  = g / BLK49;
  const int jb = g - b * BLK49;
  const float* imgB = img + (size_t)b * 3 * HW;
  float*       mB   = m_out + (size_t)b * KOUT * HW;
  _Float16* hbuf  = (_Float16*)(sScr + (size_t)wv * 16 * HSTR * 2);
  float*    msbuf = (float*)   (sScr + (size_t)wv * 16 * HSTR * 2);  // overlaid
  float*    wacc  = sAcc + wv * 256;

  for (int it = 0; it < 8; ++it) {
    const int sp0 = (jb * 8 + it) * 128 + wv * 16;

    // pixel colors (lane owns pixel lid&15)
    float x0 = imgB[sp0 + pix];
    float x1 = imgB[HW + sp0 + pix];
    float x2 = imgB[2 * HW + sp0 + pix];

    // prefetch next iteration's pixel rows (global_prefetch_b8)
    if (it < 7) {
      const int spn = sp0 + 128;
      __builtin_prefetch(&imgB[spn + pix], 0, 0);
      __builtin_prefetch(&imgB[HW + spn + pix], 0, 0);
      __builtin_prefetch(&imgB[2 * HW + spn + pix], 0, 0);
    }

    // ---- layer 1: h = relu(W1 @ x + b1), via V_WMMA_F32_16X16X4_F32 ----
    // B (4x16 f32): lanes0-15 K=0,1 ; lanes16-31 K=2,3(=pad 0)
    v2f bf; bf[0] = lo ? x0 : x2; bf[1] = lo ? x1 : 0.0f;
    #pragma unroll
    for (int mt = 0; mt < 16; ++mt) {
      const int row = mt * 16 + pix;                    // A: M = lane&15
      v2f af = *(const v2f*)((const float*)&sW1[row] + (lo ? 0 : 2)); // (w0,w1)|(w2,0)
      v8f c = {};
      c = __builtin_amdgcn_wmma_f32_16x16x4_f32(false, af, false, bf,
                                                (short)0, c, false, false);
      // element r -> hid = mt*16 + half8 + r, pixel = lane&15
      const float* bb = &sB1[mt * 16 + half8];
      union { _Float16 h[8]; uint4 u; } pk;
      #pragma unroll
      for (int r = 0; r < 8; ++r)
        pk.h[r] = (_Float16)fmaxf(c[r] + bb[r], 0.0f);
      *(uint4*)(hbuf + pix * HSTR + mt * 16 + half8) = pk.u;   // 8 consecutive halfs
    }

    // ---- layer 2: logits^T = h^T (16x256) @ W2^T (256x64), f16 WMMA ----
    // D tiles: M = pixel (regs), N = k (lanes)
    v8f acc[4] = {};
    #pragma unroll
    for (int kk = 0; kk < 8; ++kk) {
      // A fragment (h^T 16x32): lane M=pixel; j<8 -> K=kk*32+half8+j, j>=8 -> +16
      union { v16h v; uint4 u[2]; } A;
      const _Float16* hp = hbuf + pix * HSTR + kk * 32 + half8;
      A.u[0] = *(const uint4*)(hp);
      A.u[1] = *(const uint4*)(hp + 16);
      #pragma unroll
      for (int nt = 0; nt < 4; ++nt) {
        // B fragment (W2^T 32x16): lane N=k=nt*16+pix; j<8 -> K=kk*32+half8+j, j>=8 -> +16
        union { v16h v; uint4 u[2]; } Bf;
        const _Float16* wp = &sW2[(nt * 16 + pix) * W2STR + kk * 32 + half8];
        Bf.u[0] = *(const uint4*)(wp);
        Bf.u[1] = *(const uint4*)(wp + 16);
        acc[nt] = __builtin_amdgcn_wmma_f32_16x16x32_f16(false, A.v, false, Bf.v,
                                                         (short)0, acc[nt], false, false);
      }
    }

    // ---- softmax over k (per pixel): k spans 16 lanes (half) x 4 tiles ----
    float mx[8], inv[8];
    #pragma unroll
    for (int r = 0; r < 8; ++r) {
      float v = fmaxf(fmaxf(acc[0][r], acc[1][r]), fmaxf(acc[2][r], acc[3][r]));
      v = fmaxf(v, __shfl_xor(v, 1, 32));
      v = fmaxf(v, __shfl_xor(v, 2, 32));
      v = fmaxf(v, __shfl_xor(v, 4, 32));
      v = fmaxf(v, __shfl_xor(v, 8, 32));
      mx[r] = v;
    }
    #pragma unroll
    for (int nt = 0; nt < 4; ++nt)
      #pragma unroll
      for (int r = 0; r < 8; ++r)
        acc[nt][r] = __expf(acc[nt][r] - mx[r]);
    #pragma unroll
    for (int r = 0; r < 8; ++r) {
      float s = acc[0][r] + acc[1][r] + acc[2][r] + acc[3][r];
      s += __shfl_xor(s, 1, 32);
      s += __shfl_xor(s, 2, 32);
      s += __shfl_xor(s, 4, 32);
      s += __shfl_xor(s, 8, 32);
      inv[r] = __builtin_amdgcn_rcpf(s);                 // v_rcp_f32, 1 TRANS op
    }
    #pragma unroll
    for (int nt = 0; nt < 4; ++nt)
      #pragma unroll
      for (int r = 0; r < 8; ++r)
        acc[nt][r] *= inv[r];

    // per-lane copy of all 16 pixel colors indexed by reg (pixel = half8 + r)
    float xr0[8], xr1[8], xr2[8];
    #pragma unroll
    for (int r = 0; r < 8; ++r) {
      xr0[r] = __shfl(x0, half8 + r, 32);
      xr1[r] = __shfl(x1, half8 + r, 32);
      xr2[r] = __shfl(x2, half8 + r, 32);
    }

    // ---- num/den partial sums over the 16 pixels (deterministic, per-wave) ----
    #pragma unroll
    for (int nt = 0; nt < 4; ++nt) {
      float d = 0.f, n0 = 0.f, n1 = 0.f, n2 = 0.f;
      #pragma unroll
      for (int r = 0; r < 8; ++r) {
        float mv = acc[nt][r];
        d += mv;
        n0 = fmaf(xr0[r], mv, n0);
        n1 = fmaf(xr1[r], mv, n1);
        n2 = fmaf(xr2[r], mv, n2);
      }
      d  += __shfl_xor(d, 16, 32);
      n0 += __shfl_xor(n0, 16, 32);
      n1 += __shfl_xor(n1, 16, 32);
      n2 += __shfl_xor(n2, 16, 32);
      if (lo) {
        int k = nt * 16 + pix;
        wacc[k]        += d;     // den
        wacc[64  + k]  += n0;    // num c=0
        wacc[128 + k]  += n1;    // num c=1
        wacc[192 + k]  += n2;    // num c=2
      }
    }

    // ---- transpose m through LDS (overlaid on hbuf) and store coalesced ----
    #pragma unroll
    for (int nt = 0; nt < 4; ++nt) {
      float* dst = msbuf + (nt * 16 + pix) * MSTR + half8;
      *(float4*)(dst)     = make_float4(acc[nt][0], acc[nt][1], acc[nt][2], acc[nt][3]);
      *(float4*)(dst + 4) = make_float4(acc[nt][4], acc[nt][5], acc[nt][6], acc[nt][7]);
    }
    // 8 x b128: each instruction covers 8 full k-rows (512B of payload)
    #pragma unroll
    for (int j = 0; j < 8; ++j) {
      int k  = j * 8 + (lid >> 2);
      int p4 = (lid & 3) * 4;
      float4 v = *(const float4*)(msbuf + k * MSTR + p4);
      *(float4*)(mB + (size_t)k * HW + sp0 + p4) = v;
    }
  }

  // ---- block partial: fixed-order reduction of wave accumulators ----
  __syncthreads();
  float s = 0.0f;
  #pragma unroll
  for (int w = 0; w < NWAVE; ++w) s += sAcc[w * 256 + t];
  partials[(size_t)g * 256 + t] = s;
}

// ---------------------------------------------------------------------------
// Kernel 2: reduce 49 block partials per batch image -> palette
// grid = 16 blocks x 256 threads; slot t: [0..63]=den, [64+c*64+k]=num
// ---------------------------------------------------------------------------
__global__ __launch_bounds__(256) void colorcnn_palette(
    const float* __restrict__ partials, float* __restrict__ pal_out)
{
  __shared__ float red[256];
  const int b = blockIdx.x, t = threadIdx.x;
  float s = 0.0f;
  const float* p = partials + (size_t)b * BLK49 * 256 + t;
  #pragma unroll
  for (int j = 0; j < BLK49; ++j) s += p[j * 256];
  red[t] = s;
  __syncthreads();
  if (t < 192) {
    float pal = red[64 + t] / (red[t & 63] + 1e-8f);   // num/(den+eps), COLOR_NORM=1
    pal_out[b * 192 + t] = pal;
  }
}

// ---------------------------------------------------------------------------
// Kernel 3: transformed = einsum('bkhw,bck->bchw', m, palette)
// grid = 784 blocks x 256 threads, 4 pixels each (float4)
// ---------------------------------------------------------------------------
__global__ __launch_bounds__(256) void colorcnn_transform(
    const float* __restrict__ m_in, const float* __restrict__ pal,
    float* __restrict__ out)
{
  __shared__ float sP[192];
  const int t = threadIdx.x;
  const size_t p0 = ((size_t)blockIdx.x * 256 + t) * 4;
  const int b  = (int)(p0 / HW);                 // uniform per block (1024 | 50176)
  const int sp = (int)(p0 - (size_t)b * HW);
  if (t < 192) sP[t] = pal[b * 192 + t];
  __syncthreads();
  const float* mp = m_in + (size_t)b * KOUT * HW + sp;
  float4 c0 = make_float4(0.f, 0.f, 0.f, 0.f);
  float4 c1 = c0, c2 = c0;
  #pragma unroll 8
  for (int k = 0; k < KOUT; ++k) {
    if ((k & 15) == 0 && k + 16 < KOUT)
      __builtin_prefetch(mp + (size_t)(k + 16) * HW, 0, 0);
    float4 mv = *(const float4*)(mp + (size_t)k * HW);
    float p0c = sP[k], p1c = sP[64 + k], p2c = sP[128 + k];
    c0.x = fmaf(mv.x, p0c, c0.x); c0.y = fmaf(mv.y, p0c, c0.y);
    c0.z = fmaf(mv.z, p0c, c0.z); c0.w = fmaf(mv.w, p0c, c0.w);
    c1.x = fmaf(mv.x, p1c, c1.x); c1.y = fmaf(mv.y, p1c, c1.y);
    c1.z = fmaf(mv.z, p1c, c1.z); c1.w = fmaf(mv.w, p1c, c1.w);
    c2.x = fmaf(mv.x, p2c, c2.x); c2.y = fmaf(mv.y, p2c, c2.y);
    c2.z = fmaf(mv.z, p2c, c2.z); c2.w = fmaf(mv.w, p2c, c2.w);
  }
  float* o = out + (size_t)b * 3 * HW + sp;
  *(float4*)(o)          = c0;
  *(float4*)(o + HW)     = c1;
  *(float4*)(o + 2 * HW) = c2;
}

// ---------------------------------------------------------------------------
extern "C" void kernel_launch(void* const* d_in, const int* in_sizes, int n_in,
                              void* d_out, int out_size, void* d_ws, size_t ws_size,
                              hipStream_t stream)
{
  const float* img = (const float*)d_in[0];   // (16,3,224,224)
  const float* W1  = (const float*)d_in[1];   // (256,3)
  const float* b1  = (const float*)d_in[2];   // (256,)
  const float* W2  = (const float*)d_in[3];   // (64,256)

  float* out      = (float*)d_out;
  float* t_out    = out;                                   // (16,3,224,224)
  float* m_out    = out + (size_t)BATCH * 3 * HW;          // (16,64,224,224)
  float* pal_out  = m_out + (size_t)BATCH * KOUT * HW;     // (16,3,64,1,1)
  float* partials = (float*)d_ws;                          // 784*256 floats

  colorcnn_main<<<BATCH * BLK49, 256, 0, stream>>>(img, W1, b1, W2, m_out, partials);
  colorcnn_palette<<<BATCH, 256, 0, stream>>>(partials, pal_out);
  colorcnn_transform<<<(BATCH * HW) / 1024, 256, 0, stream>>>(m_out, pal_out, t_out);
}
